// AttnFPN_56564719288655
// MI455X (gfx1250) — matrix operand order repaired and checked by reading, not compiled, not run on hardware
//
#include <hip/hip_runtime.h>

// ---------------- types ----------------
typedef __attribute__((ext_vector_type(16))) __bf16 v16bf;
typedef __attribute__((ext_vector_type(8)))  float  v8f;

union AFrag { v16bf v; uint4 u[2]; };

__device__ inline __bf16 f2bf(float f) {
    union { unsigned u; float f; } c; c.f = f;
    unsigned u = c.u;
    unsigned r = u + 0x7FFFu + ((u >> 16) & 1u);
    unsigned short h = (unsigned short)(r >> 16);
    union { unsigned short s; __bf16 b; } o; o.s = h; return o.b;
}

// CDNA5 async global->LDS copy (16B per lane), tracked by ASYNCcnt.
__device__ inline void async_ld_b128(unsigned ldsOff, const void* gptr) {
    asm volatile("global_load_async_to_lds_b128 %0, %1, off"
                 :: "v"(ldsOff), "v"(gptr) : "memory");
}
__device__ inline void wait_asynccnt0() {
    asm volatile("s_wait_asynccnt 0x0" ::: "memory");
}
__device__ inline void wait_asynccnt3() {
    asm volatile("s_wait_asynccnt 0x3" ::: "memory");
}

#define BB    2
#define CIN_  512
#define COUT_ 256
#define TT    2048
#define NLY_  4
#define HH    8
#define HD_   32
#define DFF_  1024

// ---------------- elementwise helpers ----------------
__global__ void convert_kernel(const float* __restrict__ s, __bf16* __restrict__ d, int n) {
    int i = blockIdx.x * blockDim.x + threadIdx.x;
    if (i < n) d[i] = f2bf(s[i]);
}

// x (B, 512, 2048) f32 -> xT (B*2048, 512) bf16
__global__ void xt_kernel(const float* __restrict__ x, __bf16* __restrict__ xT) {
    int i = blockIdx.x * blockDim.x + threadIdx.x;
    int total = BB * TT * CIN_;
    if (i >= total) return;
    int ci = i % CIN_;
    int t  = (i / CIN_) % TT;
    int b  = i / (CIN_ * TT);
    xT[i] = f2bf(x[((size_t)b * CIN_ + ci) * TT + t]);
}

// src (B, Tin, C) bf16 -> dst (B*Tout, C*3) bf16  (stride2, pad1, k=3)
__global__ void im2col_kernel(const __bf16* __restrict__ src, __bf16* __restrict__ dst,
                              int Tin, int Tout, int C, int total) {
    int i = blockIdx.x * blockDim.x + threadIdx.x;
    if (i >= total) return;
    int kk  = i % (C * 3);
    int row = i / (C * 3);
    int ci = kk / 3, j = kk % 3;
    int to = row % Tout, b = row / Tout;
    int ti = 2 * to + j - 1;
    __bf16 v = f2bf(0.0f);
    if (ti >= 0 && ti < Tin) v = src[((size_t)b * Tin + ti) * C + ci];
    dst[i] = v;
}

// p (B*tl,256) f32 -> f = p*16 + PE ; also bf16 copy
__global__ void pe_scale_kernel(const float* __restrict__ p, float* __restrict__ f,
                                __bf16* __restrict__ fb, int tl, int total) {
    int i = blockIdx.x * blockDim.x + threadIdx.x;
    if (i >= total) return;
    int c = i % COUT_;
    int t = (i / COUT_) % tl;
    int pair = c >> 1;
    float div = __expf((float)(2 * pair) * (-9.210340371976184f / 256.0f));
    float ang = (float)t * div;
    float pe = (c & 1) ? __cosf(ang) : __sinf(ang);
    float v = p[i] * 16.0f + pe;
    f[i] = v; fb[i] = f2bf(v);
}

// f (B*tl,256) f32 -> out (B,256,1920) at time offset toff
__global__ void out_store_kernel(const float* __restrict__ f, float* __restrict__ out,
                                 int tl, int toff, int total) {
    int i = blockIdx.x * blockDim.x + threadIdx.x;
    if (i >= total) return;
    int c = i % COUT_;
    int t = (i / COUT_) % tl;
    int b = i / (COUT_ * tl);
    out[((size_t)b * COUT_ + c) * 1920 + toff + t] = f[i];
}

// memV f32 (B,2048,256) -> Vt bf16 (B,H,32,2048)
__global__ void vtrans_kernel(const float* __restrict__ v, __bf16* __restrict__ vt, int total) {
    int i = blockIdx.x * blockDim.x + threadIdx.x;
    if (i >= total) return;
    int c = i % COUT_;
    int t = (i / COUT_) % TT;
    int b = i / (COUT_ * TT);
    int h = c >> 5, d = c & 31;
    vt[(((size_t)b * HH + h) * HD_ + d) * TT + t] = f2bf(v[i]);
}

// ---------------- WMMA GEMM: Y = act(A(MxK) @ W(NxK)^T + bias) [+ res] ----------------
// Block tile 64(M) x 128(N), 8 waves in 2x4; each wave computes 32x32 -> 4 WMMAs / 32-K step.
// Tiles staged via CDNA5 global_load_async_to_lds_b128 with 2-stage LDS double-buffering:
// while WMMAs consume buffer p, the ASYNC copies for the next K-tile stream into buffer p^1.
__global__ __launch_bounds__(256) void gemm_kernel(
    const __bf16* __restrict__ A, const __bf16* __restrict__ W,
    const float* __restrict__ bias, const float* __restrict__ res,
    float* __restrict__ outF, __bf16* __restrict__ outB,
    int M, int N, int K, int relu)
{
    __shared__ __align__(16) __bf16 As[2][64 * 32];
    __shared__ __align__(16) __bf16 Ws[2][128 * 32];
    int tid  = threadIdx.x;
    int wave = tid >> 5, lane = tid & 31;
    int wm = wave >> 2;      // 0..1  (M sub-tile of 32)
    int wn = wave & 3;       // 0..3  (N sub-tile of 32)
    int h2 = lane >> 4, nl = lane & 15;
    size_t m0 = (size_t)blockIdx.y * 64;
    size_t n0 = (size_t)blockIdx.x * 128;
    // loaders: A 64x32 (8 elems/thread), W 128x32 (16 elems/thread)
    int lrowA = tid >> 2;
    int lcolA = (tid & 3) * 8;
    int lrowW = tid >> 1;
    int lcolW = (tid & 1) * 16;
    const __bf16* aPtr = &A[(m0 + lrowA) * (size_t)K + lcolA];
    const __bf16* wPtr = &W[(n0 + lrowW) * (size_t)K + lcolW];
    unsigned asOff[2], wsOff[2];
    for (int pb = 0; pb < 2; ++pb) {
        asOff[pb] = (unsigned)(size_t)&As[pb][lrowA * 32 + lcolA];
        wsOff[pb] = (unsigned)(size_t)&Ws[pb][lrowW * 32 + lcolW];
    }
    v8f acc[2][2];
    for (int mi = 0; mi < 2; ++mi)
        for (int ni = 0; ni < 2; ++ni)
            for (int r = 0; r < 8; ++r) acc[mi][ni][r] = 0.0f;
    // prologue: stage first tile into buffer 0
    async_ld_b128(asOff[0],      aPtr);
    async_ld_b128(wsOff[0],      wPtr);
    async_ld_b128(wsOff[0] + 16, wPtr + 8);
    int p = 0;
    for (int k0 = 0; k0 < K; k0 += 32) {
        int kn = k0 + 32;
        if (kn < K) {
            // stage next tile into alternate buffer while current one is consumed
            async_ld_b128(asOff[p ^ 1],      aPtr + kn);
            async_ld_b128(wsOff[p ^ 1],      wPtr + kn);
            async_ld_b128(wsOff[p ^ 1] + 16, wPtr + kn + 8);
            __builtin_prefetch(aPtr + kn + 32, 0, 1);
            __builtin_prefetch(wPtr + kn + 32, 0, 1);
            wait_asynccnt3();   // current tile's 3 copies complete; next 3 in flight
        } else {
            wait_asynccnt0();
        }
        __syncthreads();
        AFrag af[2], wf[2];
        for (int mi = 0; mi < 2; ++mi) {
            const __bf16* ar = &As[p][(wm * 32 + mi * 16 + nl) * 32];
            af[mi].u[0] = *(const uint4*)(ar + h2 * 8);
            af[mi].u[1] = *(const uint4*)(ar + 16 + h2 * 8);
        }
        for (int ni = 0; ni < 2; ++ni) {
            const __bf16* wr = &Ws[p][(wn * 32 + ni * 16 + nl) * 32];
            wf[ni].u[0] = *(const uint4*)(wr + h2 * 8);
            wf[ni].u[1] = *(const uint4*)(wr + 16 + h2 * 8);
        }
        acc[0][0] = __builtin_amdgcn_wmma_f32_16x16x32_bf16(false, af[0].v, false, wf[0].v, (short)0, acc[0][0], false, false);
        acc[0][1] = __builtin_amdgcn_wmma_f32_16x16x32_bf16(false, af[0].v, false, wf[1].v, (short)0, acc[0][1], false, false);
        acc[1][0] = __builtin_amdgcn_wmma_f32_16x16x32_bf16(false, af[1].v, false, wf[0].v, (short)0, acc[1][0], false, false);
        acc[1][1] = __builtin_amdgcn_wmma_f32_16x16x32_bf16(false, af[1].v, false, wf[1].v, (short)0, acc[1][1], false, false);
        __syncthreads();
        p ^= 1;
    }
    for (int mi = 0; mi < 2; ++mi) {
        size_t row = m0 + wm * 32 + mi * 16 + h2 * 8;
        for (int ni = 0; ni < 2; ++ni) {
            size_t col = n0 + wn * 32 + ni * 16 + nl;
            float bval = bias[col];
            for (int r = 0; r < 8; ++r) {
                size_t idx = (row + r) * (size_t)N + col;
                float v = acc[mi][ni][r] + bval;
                if (relu) v = fmaxf(v, 0.0f);
                if (res) v += res[idx];
                outF[idx] = v;
                if (outB) outB[idx] = f2bf(v);
            }
        }
    }
}

// ---------------- banded self-attention (|i-j|<=4), scalar ----------------
__global__ void selfattn_kernel(const float* __restrict__ qkv, __bf16* __restrict__ o,
                                int tl, int total) {
    int i = blockIdx.x * blockDim.x + threadIdx.x;
    if (i >= total) return;
    int q = i % tl;
    int h = (i / tl) % HH;
    int b = i / (tl * HH);
    const float* qp = qkv + ((size_t)b * tl + q) * 768 + h * HD_;
    int j0 = q - 4; if (j0 < 0) j0 = 0;
    int j1 = q + 4; if (j1 > tl - 1) j1 = tl - 1;
    float s[9];
    int cnt = j1 - j0 + 1;
    float mx = -3.0e38f;
    for (int j = 0; j < cnt; ++j) {
        const float* kp = qkv + ((size_t)b * tl + j0 + j) * 768 + 256 + h * HD_;
        float acc = 0.f;
        for (int d = 0; d < HD_; ++d) acc += qp[d] * kp[d];
        acc *= 0.17677669529663687f;
        s[j] = acc;
        mx = fmaxf(mx, acc);
    }
    float sum = 0.f;
    for (int j = 0; j < cnt; ++j) { s[j] = __expf(s[j] - mx); sum += s[j]; }
    float inv = 1.0f / sum;
    float ov[HD_];
    for (int d = 0; d < HD_; ++d) ov[d] = 0.f;
    for (int j = 0; j < cnt; ++j) {
        const float* vp = qkv + ((size_t)b * tl + j0 + j) * 768 + 512 + h * HD_;
        float p = s[j] * inv;
        for (int d = 0; d < HD_; ++d) ov[d] += p * vp[d];
    }
    __bf16* op = o + ((size_t)b * tl + q) * COUT_ + h * HD_;
    for (int d = 0; d < HD_; ++d) op[d] = f2bf(ov[d]);
}

// ---------------- fused flash cross-attention: 1 wave = 16 queries ----------------
__global__ __launch_bounds__(32) void crossattn_kernel(
    const __bf16* __restrict__ Q,   // (B, Lq, 256)
    const __bf16* __restrict__ Km,  // (B, Lk, 256)
    const __bf16* __restrict__ Vt,  // (B, H, 32, Lk)
    __bf16* __restrict__ O,         // (B, Lq, 256)
    int Lq, int Lk)
{
    __shared__ __align__(16) __bf16 Pl[16 * 32];
    int lane = threadIdx.x;
    int h2 = lane >> 4, nl = lane & 15;
    int b = blockIdx.z, h = blockIdx.y;
    int q0 = blockIdx.x * 16;
    const float scale = 0.17677669529663687f;

    AFrag aq;
    const __bf16* qb = Q + ((size_t)(b * Lq + q0 + nl)) * COUT_ + h * HD_;
    aq.u[0] = *(const uint4*)(qb + h2 * 8);
    aq.u[1] = *(const uint4*)(qb + 16 + h2 * 8);

    float mst[8], lst[8];
    for (int r = 0; r < 8; ++r) { mst[r] = -3.0e38f; lst[r] = 0.f; }
    v8f acc0 = {0.f,0.f,0.f,0.f,0.f,0.f,0.f,0.f};
    v8f acc1 = {0.f,0.f,0.f,0.f,0.f,0.f,0.f,0.f};
    v8f zero = {0.f,0.f,0.f,0.f,0.f,0.f,0.f,0.f};

    for (int kt = 0; kt < Lk; kt += 32) {
        AFrag kf0, kf1;
        const __bf16* kb0 = Km + ((size_t)(b * Lk + kt + nl)) * COUT_ + h * HD_;
        kf0.u[0] = *(const uint4*)(kb0 + h2 * 8);
        kf0.u[1] = *(const uint4*)(kb0 + 16 + h2 * 8);
        const __bf16* kb1 = Km + ((size_t)(b * Lk + kt + 16 + nl)) * COUT_ + h * HD_;
        kf1.u[0] = *(const uint4*)(kb1 + h2 * 8);
        kf1.u[1] = *(const uint4*)(kb1 + 16 + h2 * 8);
        v8f s0 = __builtin_amdgcn_wmma_f32_16x16x32_bf16(false, aq.v, false, kf0.v, (short)0, zero, false, false);
        v8f s1 = __builtin_amdgcn_wmma_f32_16x16x32_bf16(false, aq.v, false, kf1.v, (short)0, zero, false, false);

        for (int r = 0; r < 8; ++r) {
            float a = s0[r] * scale;
            float c = s1[r] * scale;
            float mx = fmaxf(a, c);
            mx = fmaxf(mx, __shfl_xor(mx, 1, 32));
            mx = fmaxf(mx, __shfl_xor(mx, 2, 32));
            mx = fmaxf(mx, __shfl_xor(mx, 4, 32));
            mx = fmaxf(mx, __shfl_xor(mx, 8, 32));
            float mnew = fmaxf(mst[r], mx);
            float corr = __expf(mst[r] - mnew);
            float p0 = __expf(a - mnew);
            float p1 = __expf(c - mnew);
            float ps = p0 + p1;
            ps += __shfl_xor(ps, 1, 32);
            ps += __shfl_xor(ps, 2, 32);
            ps += __shfl_xor(ps, 4, 32);
            ps += __shfl_xor(ps, 8, 32);
            lst[r] = lst[r] * corr + ps;
            mst[r] = mnew;
            acc0[r] *= corr;
            acc1[r] *= corr;
            int rowm = h2 * 8 + r;
            Pl[rowm * 32 + nl]      = f2bf(p0);
            Pl[rowm * 32 + 16 + nl] = f2bf(p1);
        }
        __syncthreads();
        AFrag ap;
        const __bf16* pb = &Pl[nl * 32];
        ap.u[0] = *(const uint4*)(pb + h2 * 8);
        ap.u[1] = *(const uint4*)(pb + 16 + h2 * 8);
        AFrag vf0, vf1;
        const __bf16* vb0 = Vt + (((size_t)(b * HH + h)) * HD_ + nl) * (size_t)Lk + kt;
        vf0.u[0] = *(const uint4*)(vb0 + h2 * 8);
        vf0.u[1] = *(const uint4*)(vb0 + 16 + h2 * 8);
        const __bf16* vb1 = Vt + (((size_t)(b * HH + h)) * HD_ + 16 + nl) * (size_t)Lk + kt;
        vf1.u[0] = *(const uint4*)(vb1 + h2 * 8);
        vf1.u[1] = *(const uint4*)(vb1 + 16 + h2 * 8);
        acc0 = __builtin_amdgcn_wmma_f32_16x16x32_bf16(false, ap.v, false, vf0.v, (short)0, acc0, false, false);
        acc1 = __builtin_amdgcn_wmma_f32_16x16x32_bf16(false, ap.v, false, vf1.v, (short)0, acc1, false, false);
        __syncthreads();
    }
    for (int r = 0; r < 8; ++r) {
        float inv = 1.0f / lst[r];
        size_t row = (size_t)(b * Lq + q0 + h2 * 8 + r);
        O[row * COUT_ + h * HD_ + nl]      = f2bf(acc0[r] * inv);
        O[row * COUT_ + h * HD_ + 16 + nl] = f2bf(acc1[r] * inv);
    }
}

// ---------------- layernorm over 256 cols ----------------
__global__ __launch_bounds__(256) void ln_kernel(const float* __restrict__ x,
                                                 const float* __restrict__ g,
                                                 const float* __restrict__ bt,
                                                 float* __restrict__ oF,
                                                 __bf16* __restrict__ oB) {
    __shared__ float red[256];
    int row = blockIdx.x, c = threadIdx.x;
    float v = x[(size_t)row * 256 + c];
    red[c] = v; __syncthreads();
    for (int s = 128; s > 0; s >>= 1) { if (c < s) red[c] += red[c + s]; __syncthreads(); }
    float mean = red[0] * (1.0f / 256.0f);
    __syncthreads();
    float d = v - mean;
    red[c] = d * d; __syncthreads();
    for (int s = 128; s > 0; s >>= 1) { if (c < s) red[c] += red[c + s]; __syncthreads(); }
    float var = red[0] * (1.0f / 256.0f);
    float y = d * rsqrtf(var + 1e-5f) * g[c] + bt[c];
    oF[(size_t)row * 256 + c] = y;
    oB[(size_t)row * 256 + c] = f2bf(y);
}

// ---------------- host orchestration ----------------
extern "C" void kernel_launch(void* const* d_in, const int* in_sizes, int n_in,
                              void* d_out, int out_size, void* d_ws, size_t ws_size,
                              hipStream_t stream) {
    (void)in_sizes; (void)n_in; (void)out_size; (void)ws_size;
    const float* x        = (const float*)d_in[0];
    const float* conv_w   = (const float*)d_in[1];
    const float* conv_b   = (const float*)d_in[2];
    const float* neck_w0  = (const float*)d_in[3];
    const float* neck_b0  = (const float*)d_in[4];
    const float* neck_w   = (const float*)d_in[5];
    const float* neck_b   = (const float*)d_in[6];
    const float* sa_in_w  = (const float*)d_in[7];
    const float* sa_in_b  = (const float*)d_in[8];
    const float* sa_out_w = (const float*)d_in[9];
    const float* sa_out_b = (const float*)d_in[10];
    const float* ca_in_w  = (const float*)d_in[11];
    const float* ca_in_b  = (const float*)d_in[12];
    const float* ca_out_w = (const float*)d_in[13];
    const float* ca_out_b = (const float*)d_in[14];
    const float* ff1_w    = (const float*)d_in[15];
    const float* ff1_b    = (const float*)d_in[16];
    const float* ff2_w    = (const float*)d_in[17];
    const float* ff2_b    = (const float*)d_in[18];
    const float* ln1_g    = (const float*)d_in[19];
    const float* ln1_b    = (const float*)d_in[20];
    const float* ln2_g    = (const float*)d_in[21];
    const float* ln2_b    = (const float*)d_in[22];
    const float* ln3_g    = (const float*)d_in[23];
    const float* ln3_b    = (const float*)d_in[24];
    float* out = (float*)d_out;

    char* base = (char*)d_ws;
    size_t off = 0;
    auto alloc = [&](size_t bytes) -> char* {
        char* p = base + off;
        off += (bytes + 255) & ~(size_t)255;
        return p;
    };
    __bf16* xT     = (__bf16*)alloc((size_t)2 * 2048 * 512 * 2);
    __bf16* wConv  = (__bf16*)alloc((size_t)256 * 512 * 2);
    __bf16* wNeck0 = (__bf16*)alloc((size_t)256 * 1536 * 2);
    __bf16* wNeck  = (__bf16*)alloc((size_t)3 * 256 * 768 * 2);
    __bf16* wSaIn  = (__bf16*)alloc((size_t)4 * 768 * 256 * 2);
    __bf16* wSaOut = (__bf16*)alloc((size_t)4 * 256 * 256 * 2);
    __bf16* wCaIn  = (__bf16*)alloc((size_t)4 * 768 * 256 * 2);
    __bf16* wCaOut = (__bf16*)alloc((size_t)4 * 256 * 256 * 2);
    __bf16* wFf1   = (__bf16*)alloc((size_t)4 * 1024 * 256 * 2);
    __bf16* wFf2   = (__bf16*)alloc((size_t)4 * 256 * 1024 * 2);
    __bf16* memBf  = (__bf16*)alloc((size_t)4096 * 256 * 2);
    __bf16* memK   = (__bf16*)alloc((size_t)4 * 4096 * 256 * 2);
    __bf16* VtAll  = (__bf16*)alloc((size_t)4 * 2 * 8 * 32 * 2048 * 2);
    float*  vtmp   = (float*)alloc((size_t)4096 * 256 * 4);
    float*  dump   = (float*)alloc((size_t)2048 * 1024 * 4);
    __bf16* Acol   = (__bf16*)alloc((size_t)2048 * 1536 * 2);
    int TL[4]   = {1024, 512, 256, 128};
    int TOFF[4] = {0, 1024, 1536, 1792};
    float*  pF[4]; __bf16* pB[4];
    for (int i = 0; i < 4; ++i) {
        pF[i] = (float*)alloc((size_t)2 * TL[i] * 256 * 4);
        pB[i] = (__bf16*)alloc((size_t)2 * TL[i] * 256 * 2);
    }
    float*  fBuf  = (float*)alloc((size_t)2048 * 256 * 4);
    __bf16* fBf   = (__bf16*)alloc((size_t)2048 * 256 * 2);
    float*  t1    = (float*)alloc((size_t)2048 * 256 * 4);
    float*  qkv   = (float*)alloc((size_t)2048 * 768 * 4);
    __bf16* attnB = (__bf16*)alloc((size_t)2048 * 256 * 2);
    __bf16* qB    = (__bf16*)alloc((size_t)2048 * 256 * 2);
    __bf16* caB   = (__bf16*)alloc((size_t)2048 * 256 * 2);
    __bf16* h1B   = (__bf16*)alloc((size_t)2048 * 1024 * 2);

    auto cvt = [&](const float* s, __bf16* d, int n) {
        convert_kernel<<<(n + 255) / 256, 256, 0, stream>>>(s, d, n);
    };
    auto gemm = [&](const __bf16* A, const __bf16* W, const float* bias, const float* res,
                    float* oF, __bf16* oB, int M, int N, int K, int relu) {
        dim3 g(N / 128, M / 64);
        gemm_kernel<<<g, 256, 0, stream>>>(A, W, bias, res, oF, oB, M, N, K, relu);
    };

    // weight conversion
    cvt(conv_w,   wConv,  256 * 512);
    cvt(neck_w0,  wNeck0, 256 * 1536);
    cvt(neck_w,   wNeck,  3 * 256 * 768);
    cvt(sa_in_w,  wSaIn,  4 * 768 * 256);
    cvt(sa_out_w, wSaOut, 4 * 256 * 256);
    cvt(ca_in_w,  wCaIn,  4 * 768 * 256);
    cvt(ca_out_w, wCaOut, 4 * 256 * 256);
    cvt(ff1_w,    wFf1,   4 * 1024 * 256);
    cvt(ff2_w,    wFf2,   4 * 256 * 1024);

    // x transpose + memory (1x1 conv, relu)
    {
        int n = 2 * 2048 * 512;
        xt_kernel<<<(n + 255) / 256, 256, 0, stream>>>(x, xT);
    }
    gemm(xT, wConv, conv_b, nullptr, dump, memBf, 4096, 256, 512, 1);

    // pyramid
    {
        int total = 2 * 1024 * 1536;
        im2col_kernel<<<(total + 255) / 256, 256, 0, stream>>>(xT, Acol, 2048, 1024, 512, total);
        gemm(Acol, wNeck0, neck_b0, nullptr, pF[0], pB[0], 2048, 256, 1536, 1);
    }
    for (int i = 1; i < 4; ++i) {
        int Tin = TL[i - 1], Tout = TL[i];
        int total = 2 * Tout * 768;
        im2col_kernel<<<(total + 255) / 256, 256, 0, stream>>>(pB[i - 1], Acol, Tin, Tout, 256, total);
        gemm(Acol, wNeck + (size_t)(i - 1) * 256 * 768, neck_b + (i - 1) * 256,
             nullptr, pF[i], pB[i], 2 * Tout, 256, 768, 1);
    }

    // cross-attn K/V of memory, per layer (shared across levels)
    for (int l = 0; l < 4; ++l) {
        gemm(memBf, wCaIn + (size_t)l * 768 * 256 + 256 * 256, ca_in_b + l * 768 + 256,
             nullptr, dump, memK + (size_t)l * 4096 * 256, 4096, 256, 256, 0);
        gemm(memBf, wCaIn + (size_t)l * 768 * 256 + 512 * 256, ca_in_b + l * 768 + 512,
             nullptr, vtmp, nullptr, 4096, 256, 256, 0);
        int total = 2 * 2048 * 256;
        vtrans_kernel<<<(total + 255) / 256, 256, 0, stream>>>(vtmp, VtAll + (size_t)l * 1048576, total);
    }

    // per-level transformer stack
    for (int lv = 0; lv < 4; ++lv) {
        int tl = TL[lv];
        int M = 2 * tl;
        int tot = M * 256;
        pe_scale_kernel<<<(tot + 255) / 256, 256, 0, stream>>>(pF[lv], fBuf, fBf, tl, tot);
        for (int l = 0; l < 4; ++l) {
            // self-attention
            gemm(fBf, wSaIn + (size_t)l * 768 * 256, sa_in_b + l * 768,
                 nullptr, qkv, nullptr, M, 768, 256, 0);
            {
                int total = 2 * 8 * tl;
                selfattn_kernel<<<(total + 255) / 256, 256, 0, stream>>>(qkv, attnB, tl, total);
            }
            gemm(attnB, wSaOut + (size_t)l * 256 * 256, sa_out_b + l * 256,
                 fBuf, t1, nullptr, M, 256, 256, 0);
            ln_kernel<<<M, 256, 0, stream>>>(t1, ln1_g + l * 256, ln1_b + l * 256, fBuf, fBf);
            // cross-attention
            gemm(fBf, wCaIn + (size_t)l * 768 * 256, ca_in_b + l * 768,
                 nullptr, dump, qB, M, 256, 256, 0);
            {
                dim3 g(tl / 16, 8, 2);
                crossattn_kernel<<<g, 32, 0, stream>>>(qB, memK + (size_t)l * 4096 * 256,
                                                       VtAll + (size_t)l * 1048576, caB, tl, 2048);
            }
            gemm(caB, wCaOut + (size_t)l * 256 * 256, ca_out_b + l * 256,
                 fBuf, t1, nullptr, M, 256, 256, 0);
            ln_kernel<<<M, 256, 0, stream>>>(t1, ln2_g + l * 256, ln2_b + l * 256, fBuf, fBf);
            // FFN
            gemm(fBf, wFf1 + (size_t)l * 1024 * 256, ff1_b + l * 1024,
                 nullptr, dump, h1B, M, 1024, 256, 1);
            gemm(h1B, wFf2 + (size_t)l * 256 * 1024, ff2_b + l * 256,
                 fBuf, t1, nullptr, M, 256, 1024, 0);
            ln_kernel<<<M, 256, 0, stream>>>(t1, ln3_g + l * 256, ln3_b + l * 256, fBuf, fBf);
        }
        out_store_kernel<<<(tot + 255) / 256, 256, 0, stream>>>(fBuf, out, tl, TOFF[lv], tot);
    }
}
